// S4D_29703993819395
// MI455X (gfx1250) — compile-verified, matrix operand order checked by reading, not run on hardware
//
#include <hip/hip_runtime.h>
#include <hip/hip_bf16.h>
#include <cstdint>

// ---------------------------------------------------------------------------
// S4D on MI455X (gfx1250, wave32, WMMA + Tensor Data Mover).
//   B=4, L=2048, H=1024, N2=32 states, chunk T=64, NCOL = B*(L/T) = 128.
// Pipeline:
//   K0a: transpose u (B,L,H) f32 -> UT[h][col=128][m=64] bf16
//   K0b: W_out f32 -> bf16
//   K1 : per-h SSM prep (one wave32, lane = state): param block PB[h] =
//        { Toeplitz T 64x64 | P (z^{63-m}) 64x64 | G (Cw z^{j+1}) 64x64 } bf16
//        + Z64 = z^64 (f32).
//   K2 : per-h fused chunk kernel (8 waves): TDM-stage PB[h]+UT[h] into LDS,
//        S = P x U (WMMA bf16), 32-step cross-chunk state scan (exact, fp32),
//        Y = G x X + T x U (WMMA), + D*u skip, exact GELU -> act bf16.
//        All WMMA B operands stored row-major [n][k] in LDS => ds_load_b128.
//   K3 : out = act(8192x1024) x W^T + b. TDM double-buffered 128x128 tiles,
//        s_wait_tensorcnt pipelining, fp32 accumulate.
// ---------------------------------------------------------------------------

#define Bq 4
#define Lq 2048
#define Hq 1024
#define N2q 32
#define NCHUNK 32      // Lq / 64
#define NCOL 128       // Bq * NCHUNK

typedef __attribute__((ext_vector_type(16))) __bf16 v16bf;
typedef __attribute__((ext_vector_type(8)))  float  v8f;
typedef unsigned int v4u  __attribute__((ext_vector_type(4)));
typedef int          v8i_ __attribute__((ext_vector_type(8)));
typedef int          v4i_ __attribute__((ext_vector_type(4)));

// --- TDM: 2-D tile load global->LDS (D# per CDNA5 ISA 8.3/8.4) -------------
// data_size=1B; tile = row_bytes x rows; row pitch = stride_bytes.
__device__ __forceinline__ void tdm_load_2d(uint32_t lds_off, const void* gaddr,
                                            uint32_t row_bytes, uint32_t rows,
                                            uint32_t stride_bytes) {
  uint64_t ga = (uint64_t)(uintptr_t)gaddr;
  v4u g0;
  g0[0] = 1u;                                        // count=1 (valid), user mode
  g0[1] = lds_off;                                   // lds_addr (bytes)
  g0[2] = (uint32_t)(ga & 0xffffffffu);              // global_addr[31:0]
  g0[3] = (uint32_t)((ga >> 32) & 0x01ffffffu) | (2u << 30);  // addr[56:32], type=2
  v8i_ g1;
  g1[0] = 0;                                         // wgmask=0, data_size=0 (1B)
  g1[1] = (int)((row_bytes & 0xffffu) << 16);        // tensor_dim0[15:0]
  g1[2] = (int)(((row_bytes >> 16) & 0xffffu) |      // tensor_dim0[31:16]
                ((rows & 0xffffu) << 16));           // tensor_dim1[15:0]
  g1[3] = (int)(((rows >> 16) & 0xffffu) |           // tensor_dim1[31:16]
                ((row_bytes & 0xffffu) << 16));      // tile_dim0
  g1[4] = (int)(rows & 0xffffu);                     // tile_dim1 (tile_dim2=0)
  g1[5] = (int)stride_bytes;                         // tensor_dim0_stride[31:0]
  g1[6] = 0;                                         // stride hi, dim1_stride lo
  g1[7] = 0;
  v4i_ gz = {0, 0, 0, 0};
#if __clang_major__ >= 23
  v8i_ gz8 = {0, 0, 0, 0, 0, 0, 0, 0};
  __builtin_amdgcn_tensor_load_to_lds(g0, g1, gz, gz, gz8, 0);
#else
  __builtin_amdgcn_tensor_load_to_lds(g0, g1, gz, gz, 0);
#endif
}

// --- WMMA fragment loader (16x16x32 bf16), row-major source ----------------
// lane r = lane&15 selects row (A: M-row, B: N-row); elem i<8 -> k=half*8+i,
// i>=8 -> k=16+half*8+(i-8). Two 16B LDS reads per lane.
__device__ __forceinline__ v16bf frag_rowmajor(const __bf16* M, int ld,
                                               int row0, int col0, int lane) {
  int r = lane & 15, hf = (lane >> 4) & 1;
  const __bf16* p = M + (row0 + r) * ld + col0;
  v16bf f;
#pragma unroll
  for (int i = 0; i < 8; ++i) f[i] = p[hf * 8 + i];
#pragma unroll
  for (int i = 0; i < 8; ++i) f[8 + i] = p[16 + hf * 8 + i];
  return f;
}

__device__ __forceinline__ float gelu_exact(float x) {
  return 0.5f * x * (1.0f + erff(x * 0.70710678118654752f));
}

// --- K0a: u (B,L,H) f32 -> UT[h][b*32+c][m] bf16 ---------------------------
__global__ __launch_bounds__(256) void s4d_transpose(const float* __restrict__ u,
                                                     __bf16* __restrict__ UT) {
  __shared__ float tile[64][65];
  int bc = blockIdx.x;
  int b = bc >> 5, c = bc & 31;
  int h0 = blockIdx.y * 64;
  int tid = threadIdx.x;
#pragma unroll
  for (int q = 0; q < 16; ++q) {
    int e = tid + q * 256;
    int hh = e & 63, m = e >> 6;                       // coalesced along h
    tile[m][hh] = u[((size_t)(b * Lq + c * 64 + m)) * Hq + h0 + hh];
  }
  __syncthreads();
#pragma unroll
  for (int q = 0; q < 16; ++q) {
    int e = tid + q * 256;
    int m = e & 63, hh = e >> 6;                       // coalesced along m
    UT[(((size_t)(h0 + hh)) * NCOL + (b * 32 + c)) * 64 + m] = (__bf16)tile[m][hh];
  }
}

// --- K0b: W_out -> bf16 ----------------------------------------------------
__global__ __launch_bounds__(256) void s4d_cvt_w(const float* __restrict__ W,
                                                 __bf16* __restrict__ Wbf, int n) {
  int i = blockIdx.x * 256 + threadIdx.x;
  if (i < n) Wbf[i] = (__bf16)W[i];
}

// --- K1: per-h SSM prep; one wave32, lane = state n ------------------------
__global__ __launch_bounds__(32) void s4d_prep(const float* __restrict__ log_dt,
                                               const float* __restrict__ log_w_real,
                                               const float* __restrict__ w_imag,
                                               const float* __restrict__ C_real,
                                               const float* __restrict__ C_imag,
                                               __bf16* __restrict__ PB,
                                               float* __restrict__ Z64) {
  __shared__ float karr[64];
  int h = blockIdx.x, n = threadIdx.x;
  float dt = __expf(log_dt[h]);
  float wr = -__expf(log_w_real[n]);
  float wi = w_imag[n];
  float cr = C_real[h * N2q + n], ci = C_imag[h * N2q + n];
  float ar = wr * dt, ai = wi * dt;
  float ea = __expf(ar);
  float zr = ea * __cosf(ai), zi = ea * __sinf(ai);    // z = exp(dt*w)
  // Cw = C * (z - 1) / w
  float er = zr - 1.0f, ei = zi;
  float inv = 1.0f / (wr * wr + wi * wi);
  float qr = (er * wr + ei * wi) * inv, qi = (ei * wr - er * wi) * inv;
  float cwr = cr * qr - ci * qi, cwi = cr * qi + ci * qr;

  __bf16* Th = PB + (size_t)h * 12288;                 // Toeplitz 64x64
  __bf16* Ph = Th + 4096;                              // P 64x64
  __bf16* Gh = Th + 8192;                              // G 64x64

  float pr = 1.0f, pi = 0.0f;                          // z^d, d=0
  for (int d = 0; d < 64; ++d) {
    float s = cwr * pr - cwi * pi;                     // k[d] butterfly reduce
    for (int off = 16; off >= 1; off >>= 1) s += __shfl_xor(s, off);
    if (n == 0) karr[d] = 2.0f * s;
    Ph[(2 * n) * 64 + (63 - d)]     = (__bf16)pr;      // P[2n][63-d] = Re z^d
    Ph[(2 * n + 1) * 64 + (63 - d)] = (__bf16)pi;
    float nr = pr * zr - pi * zi, ni = pr * zi + pi * zr;   // z^{d+1}
    Gh[d * 64 + 2 * n]     = (__bf16)(2.0f * (cwr * nr - cwi * ni));
    Gh[d * 64 + 2 * n + 1] = (__bf16)(-2.0f * (cwr * ni + cwi * nr));
    pr = nr; pi = ni;
  }
  Z64[h * 64 + 2 * n]     = pr;                        // z^64
  Z64[h * 64 + 2 * n + 1] = pi;
  __syncthreads();
  for (int j = 0; j < 64; ++j) {                       // T[j][m] = k[j-m]
    int m0 = 2 * n, m1 = 2 * n + 1;
    Th[j * 64 + m0] = (__bf16)((j >= m0) ? karr[j - m0] : 0.0f);
    Th[j * 64 + m1] = (__bf16)((j >= m1) ? karr[j - m1] : 0.0f);
  }
}

// --- K2: per-h fused chunk kernel (256 threads = 8 waves) ------------------
__global__ __launch_bounds__(256) void s4d_chunk(const __bf16* __restrict__ PB,
                                                 const __bf16* __restrict__ UT,
                                                 const float* __restrict__ Z64,
                                                 const float* __restrict__ D,
                                                 __bf16* __restrict__ act) {
  extern __shared__ char smem[];
  __bf16* Tm = (__bf16*)smem;        // 64x64   (A: Toeplitz)
  __bf16* Pm = Tm + 4096;            // 64x64   (A: chunk-input)
  __bf16* Gm = Pm + 4096;            // 64x64   (A: state->output)
  __bf16* Um = Gm + 4096;            // [col=128][m=64]   (B operand)
  __bf16* Xt = Um + 8192;            // [col=128][2n=64]  (B operand)
  float*  Sm = (float*)(Xt + 8192);  // [row=64][col=128] fp32 chunk sums

  int h = blockIdx.x, tid = threadIdx.x;
  int wave = tid >> 5, lane = tid & 31;
  int colb = wave * 16;
  int hf = (lane >> 4) & 1, n16 = lane & 15;

  // TDM-stage the per-h param block (24 KB) and U block (16 KB) into LDS.
  if (wave == 0) {
    tdm_load_2d((uint32_t)(uintptr_t)Tm, PB + (size_t)h * 12288, 24576u, 1u, 24576u);
    tdm_load_2d((uint32_t)(uintptr_t)Um, UT + (size_t)h * 8192, 16384u, 1u, 16384u);
    __builtin_amdgcn_s_wait_tensorcnt(0);
  }
  __syncthreads();

  v8f vzero = {0.f, 0.f, 0.f, 0.f, 0.f, 0.f, 0.f, 0.f};

  // GEMM1: S(64x128) = P(64x64) x U(64x128)   [B tiles from Um rows=cols]
  {
    v8f accS[4];
#pragma unroll
    for (int r = 0; r < 4; ++r) accS[r] = vzero;
#pragma unroll
    for (int kk = 0; kk < 64; kk += 32) {
      v16bf bu = frag_rowmajor(Um, 64, colb, kk, lane);
#pragma unroll
      for (int r = 0; r < 4; ++r) {
        v16bf a = frag_rowmajor(Pm, 64, r * 16, kk, lane);
        accS[r] = __builtin_amdgcn_wmma_f32_16x16x32_bf16(
            false, a, false, bu, (short)0, accS[r], false, false);
      }
    }
#pragma unroll
    for (int r = 0; r < 4; ++r)
#pragma unroll
      for (int i = 0; i < 8; ++i)
        Sm[(r * 16 + i + 8 * hf) * 128 + colb + n16] = accS[r][i];
  }
  __syncthreads();

  // cross-chunk state scan: x_c = z^64 * x_{c-1} + S[:,c];  Xt holds x_{c-1}
  if (tid < 128) {
    int b = tid >> 5, n = tid & 31;
    float zr = Z64[h * 64 + 2 * n], zi = Z64[h * 64 + 2 * n + 1];
    float xr = 0.0f, xi = 0.0f;
    uint32_t* Xw = (uint32_t*)Xt;
    for (int c = 0; c < NCHUNK; ++c) {
      int col = b * 32 + c;
      uint32_t pk = ((uint32_t)__builtin_bit_cast(unsigned short, (__bf16)xi) << 16) |
                    (uint32_t)__builtin_bit_cast(unsigned short, (__bf16)xr);
      Xw[col * 32 + n] = pk;                           // Xt[col][2n],[2n+1]
      float sr = Sm[(2 * n) * 128 + col], si = Sm[(2 * n + 1) * 128 + col];
      float nxr = zr * xr - zi * xi + sr;
      float nxi = zr * xi + zi * xr + si;
      xr = nxr; xi = nxi;
    }
  }
  __syncthreads();

  // GEMM2+3: Y = G x X + T x U ; + D*u skip, exact GELU -> act bf16
  v8f accY[4];
#pragma unroll
  for (int r = 0; r < 4; ++r) accY[r] = vzero;
#pragma unroll
  for (int kk = 0; kk < 64; kk += 32) {
    v16bf bx = frag_rowmajor(Xt, 64, colb, kk, lane);
    v16bf bu = frag_rowmajor(Um, 64, colb, kk, lane);
#pragma unroll
    for (int r = 0; r < 4; ++r) {
      v16bf ag = frag_rowmajor(Gm, 64, r * 16, kk, lane);
      accY[r] = __builtin_amdgcn_wmma_f32_16x16x32_bf16(
          false, ag, false, bx, (short)0, accY[r], false, false);
      v16bf at = frag_rowmajor(Tm, 64, r * 16, kk, lane);
      accY[r] = __builtin_amdgcn_wmma_f32_16x16x32_bf16(
          false, at, false, bu, (short)0, accY[r], false, false);
    }
  }
  float Dh = D[h];
#pragma unroll
  for (int r = 0; r < 4; ++r) {
#pragma unroll
    for (int i = 0; i < 8; ++i) {
      int row = r * 16 + i + 8 * hf;                   // j within chunk
      int col = colb + n16;                            // (b,c)
      float uval = (float)Um[col * 64 + row];
      float y = accY[r][i] + Dh * uval;
      float g = gelu_exact(y);
      int bb = col >> 5, cc = col & 31;
      act[((size_t)(bb * Lq + cc * 64 + row)) * Hq + h] = (__bf16)g;
    }
  }
}

// --- K3: out = act x W^T + b ; TDM double-buffered 128x128 tiles -----------
__global__ __launch_bounds__(256) void s4d_out_gemm(const __bf16* __restrict__ act,
                                                    const __bf16* __restrict__ Wbf,
                                                    const float* __restrict__ bias,
                                                    float* __restrict__ out) {
  __shared__ __attribute__((aligned(16))) __bf16 Al[2][128 * 32];  // tokens x k
  __shared__ __attribute__((aligned(16))) __bf16 Bl[2][128 * 32];  // feat x k
  int tid = threadIdx.x, wave = tid >> 5, lane = tid & 31;
  int col0 = blockIdx.x * 128;       // output feature tile
  int row0 = blockIdx.y * 128;       // token tile
  const int KB = Hq / 32;
  const uint32_t pitch = (uint32_t)(Hq * 2);

  v8f vzero = {0.f, 0.f, 0.f, 0.f, 0.f, 0.f, 0.f, 0.f};
  v8f acc[8];
#pragma unroll
  for (int ct = 0; ct < 8; ++ct) acc[ct] = vzero;

  if (wave == 0) {                                     // prime buffer 0
    tdm_load_2d((uint32_t)(uintptr_t)&Al[0][0], act + (size_t)row0 * Hq, 64u, 128u, pitch);
    tdm_load_2d((uint32_t)(uintptr_t)&Bl[0][0], Wbf + (size_t)col0 * Hq, 64u, 128u, pitch);
  }
  for (int kb = 0; kb < KB; ++kb) {
    int cur = kb & 1;
    if (wave == 0) {
      if (kb + 1 < KB) {                               // stream next K-slab
        int k0n = (kb + 1) * 32;
        tdm_load_2d((uint32_t)(uintptr_t)&Al[cur ^ 1][0],
                    act + (size_t)row0 * Hq + k0n, 64u, 128u, pitch);
        tdm_load_2d((uint32_t)(uintptr_t)&Bl[cur ^ 1][0],
                    Wbf + (size_t)col0 * Hq + k0n, 64u, 128u, pitch);
        __builtin_amdgcn_s_wait_tensorcnt(2);          // current pair landed
      } else {
        __builtin_amdgcn_s_wait_tensorcnt(0);
      }
    }
    __syncthreads();                                   // publish current buffer
    v16bf a = frag_rowmajor(&Al[cur][0], 32, wave * 16, 0, lane);
#pragma unroll
    for (int ct = 0; ct < 8; ++ct) {
      v16bf b = frag_rowmajor(&Bl[cur][0], 32, ct * 16, 0, lane);
      acc[ct] = __builtin_amdgcn_wmma_f32_16x16x32_bf16(
          false, a, false, b, (short)0, acc[ct], false, false);
    }
    __syncthreads();                                   // release before overwrite
  }
  int hf = (lane >> 4) & 1, n16 = lane & 15;
#pragma unroll
  for (int ct = 0; ct < 8; ++ct) {
#pragma unroll
    for (int i = 0; i < 8; ++i) {
      int row = row0 + wave * 16 + i + 8 * hf;
      int col = col0 + ct * 16 + n16;
      out[(size_t)row * Hq + col] = acc[ct][i] + bias[col];
    }
  }
}

// ---------------------------------------------------------------------------
extern "C" void kernel_launch(void* const* d_in, const int* in_sizes, int n_in,
                              void* d_out, int out_size, void* d_ws, size_t ws_size,
                              hipStream_t stream) {
  (void)in_sizes; (void)n_in; (void)out_size; (void)ws_size;
  const float* u           = (const float*)d_in[0];
  const float* log_dt      = (const float*)d_in[1];
  const float* log_w_real  = (const float*)d_in[2];
  const float* w_imag      = (const float*)d_in[3];
  const float* C_real      = (const float*)d_in[4];
  const float* C_imag      = (const float*)d_in[5];
  const float* D           = (const float*)d_in[6];
  const float* W_out       = (const float*)d_in[7];
  const float* b_out       = (const float*)d_in[8];
  float* out = (float*)d_out;

  char* ws = (char*)d_ws;
  __bf16* UT  = (__bf16*)(ws);                       // 16 MB  H*128*64 bf16
  __bf16* PB  = (__bf16*)(ws + (16u << 20));         // 24 MB  H*12288 bf16 (T|P|G)
  float*  Z64 = (float*)(ws + (40u << 20));          // 256 KB H*64 f32
  __bf16* Wbf = (__bf16*)(ws + (41u << 20));         //  2 MB
  __bf16* act = (__bf16*)(ws + (44u << 20));         // 16 MB  8192*1024 bf16

  s4d_transpose<<<dim3(Bq * NCHUNK, Hq / 64), 256, 0, stream>>>(u, UT);
  s4d_cvt_w<<<(Hq * Hq + 255) / 256, 256, 0, stream>>>(W_out, Wbf, Hq * Hq);
  s4d_prep<<<Hq, 32, 0, stream>>>(log_dt, log_w_real, w_imag, C_real, C_imag,
                                  PB, Z64);
  size_t shbytes = (size_t)(4096 * 3 + 8192 * 2) * sizeof(__bf16) + 8192 * sizeof(float);
  s4d_chunk<<<Hq, 256, shbytes, stream>>>(PB, UT, Z64, D, act);
  s4d_out_gemm<<<dim3(Hq / 128, (Bq * Lq) / 128), 256, 0, stream>>>(act, Wbf, b_out, out);
}
